// VectorQuantizer_15771119911145
// MI455X (gfx1250) — matrix operand order repaired and checked by reading, not compiled, not run on hardware
//
#include <hip/hip_runtime.h>
#include <hip/hip_bf16.h>

// CDNA5 / gfx1250 vector-quantizer forward.
// Round 2: codebook pre-split into bf16 hi/lo and pre-swizzled into the WMMA
// B-operand register layout (one-time pack kernel), so the hot loop is pure
// {vector load -> v_wmma_f32_16x16x32_bf16 -> argmin} with no conversion VALU.
// Each wave owns 32 rows (two A register sets) to double B reuse per load.

typedef __bf16 v16bf __attribute__((ext_vector_type(16)));
typedef float  v8f   __attribute__((ext_vector_type(8)));

#define D_DIM   256
#define K_CODES 4096
#define PACK_ELEMS (D_DIM * K_CODES)          // 1,048,576 elements

__device__ __forceinline__ unsigned short f2bf_rne(float f) {
  unsigned u = __float_as_uint(f);
  unsigned r = u + 0x7FFFu + ((u >> 16) & 1u);   // round-to-nearest-even
  return (unsigned short)(r >> 16);
}
__device__ __forceinline__ float bfbits2f(unsigned short b) {
  return __uint_as_float(((unsigned)b) << 16);
}
__device__ __forceinline__ __bf16 mkbf(unsigned short b) {
  return __builtin_bit_cast(__bf16, b);
}

// ---------------------------------------------------------------------------
// Kernel 0: split codebook into bf16 hi/lo, swizzled into WMMA B layout.
// Target layout: for column-tile ct (16 codes), k-chunk c (32 feats), lane L
// (col = 16*ct + L%16, K = 32*c + (L/16)*16 + e):
//   pos = ((ct*8 + c)*32 + L)*16 + e   -> lane reads 16 contiguous bf16 (32B).
// ---------------------------------------------------------------------------
__global__ void vq_pack_kernel(const float* __restrict__ emb,
                               unsigned short* __restrict__ bhi,
                               unsigned short* __restrict__ blo) {
  int idx = blockIdx.x * blockDim.x + threadIdx.x;
  if (idx >= PACK_ELEMS) return;
  const int d = idx >> 12;        // / K_CODES
  const int k = idx & (K_CODES - 1);
  const float f = emb[idx];
  const unsigned short hb = f2bf_rne(f);
  const unsigned short lb = f2bf_rne(f - bfbits2f(hb));
  const int ct = k >> 4, n = k & 15;
  const int c  = d >> 5, r = d & 31;
  const int lane = (r >> 4) * 16 + n;
  const int e    = r & 15;
  const size_t pos = ((size_t)(ct * 8 + c) * 32 + lane) * 16 + e;
  bhi[pos] = hb;
  blo[pos] = lb;
}

// ---------------------------------------------------------------------------
// Kernel 1: per-code squared norms ||e_k||^2, and zero the usage histogram.
// ---------------------------------------------------------------------------
__global__ void vq_norms_kernel(const float* __restrict__ emb,
                                float* __restrict__ esq,
                                float* __restrict__ counts) {
  int k = blockIdx.x * blockDim.x + threadIdx.x;
  if (k >= K_CODES) return;
  float s = 0.0f;
  for (int d = 0; d < D_DIM; ++d) {
    float e = emb[(size_t)d * K_CODES + k];
    s += e * e;
  }
  esq[k]    = s;
  counts[k] = 0.0f;
}

// ---------------------------------------------------------------------------
// Kernel 2: fused WMMA distance + argmin + codebook gather + histogram.
// One wave32 per 32-row strip (two 16-row WMMA tiles sharing B loads);
// 8 waves (256 threads) per block. dist surrogate = ||e||^2 - 2*(z.e).
// ---------------------------------------------------------------------------
__global__ void __launch_bounds__(256)
vq_wmma_kernel(const float* __restrict__ x,
               const unsigned short* __restrict__ bhi_pack,
               const unsigned short* __restrict__ blo_pack,
               const float* __restrict__ esq,
               const float* __restrict__ emb,
               float* __restrict__ out,
               float* __restrict__ counts,
               int n_rows) {
  const int lane = threadIdx.x & 31;
  const int wave = threadIdx.x >> 5;
  const int n    = lane & 15;   // WMMA B/C column within tile
  const int h    = lane >> 4;   // half-wave id
  const int rowBase = (blockIdx.x * 8 + wave) * 32;
  if (rowBase >= n_rows) return;   // wave-uniform exit (EXEC all-1 for WMMA)

  // ---- Load A (2 sub-tiles x 16 rows x 256 feats) as split bf16 in regs ---
  // A 16x32 k order chosen here (not AMD table order): M = lane%16; elem e -> K = h*16+e (matches packed B).
  v16bf Ahi[2][8], Alo[2][8];
  #pragma unroll
  for (int t = 0; t < 2; ++t) {
    const float* xr = x + (size_t)(rowBase + t * 16 + n) * D_DIM;
    #pragma unroll
    for (int c = 0; c < 8; ++c) {
      const int d0 = c * 32 + h * 16;
      #pragma unroll
      for (int j = 0; j < 8; ++j) {
        float f0 = xr[d0 + j];
        unsigned short h0 = f2bf_rne(f0);
        Ahi[t][c][j]     = mkbf(h0);
        Alo[t][c][j]     = mkbf(f2bf_rne(f0 - bfbits2f(h0)));
        float f1 = xr[d0 + 8 + j];
        unsigned short h1 = f2bf_rne(f1);
        Ahi[t][c][8 + j] = mkbf(h1);
        Alo[t][c][8 + j] = mkbf(f2bf_rne(f1 - bfbits2f(h1)));
      }
    }
  }

  // Running argmin. C layout: acc[v] at lane L = sim[M = v + 8*h][N = n].
  float best[2][8];
  int   bidx[2][8];
  #pragma unroll
  for (int t = 0; t < 2; ++t)
    #pragma unroll
    for (int v = 0; v < 8; ++v) { best[t][v] = 3.4e38f; bidx[t][v] = 0; }

  const v16bf* __restrict__ bh = ((const v16bf*)bhi_pack) + lane;
  const v16bf* __restrict__ bl = ((const v16bf*)blo_pack) + lane;

  for (int ct = 0; ct < K_CODES / 16; ++ct) {
    const v16bf* bht = bh + ct * 256;   // 256 v16bf per tile (8 chunks x 32 lanes)
    const v16bf* blt = bl + ct * 256;
    if (ct + 1 < K_CODES / 16) {        // prefetch next tile (global_prefetch_b8)
      __builtin_prefetch(bht + 256, 0, 0);
      __builtin_prefetch(blt + 256, 0, 0);
    }

    v8f acc0 = {}, acc1 = {};
    #pragma unroll
    for (int c = 0; c < 8; ++c) {
      const v16bf Bhi = bht[c * 32];    // 32B contiguous per lane, no VALU
      const v16bf Blo = blt[c * 32];
      // split-bf16: a.b ~= ahi*bhi + ahi*blo + alo*bhi  (error ~2^-18)
      acc0 = __builtin_amdgcn_wmma_f32_16x16x32_bf16(false, Ahi[0][c], false, Bhi,
                                                     (short)0, acc0, false, false);
      acc0 = __builtin_amdgcn_wmma_f32_16x16x32_bf16(false, Ahi[0][c], false, Blo,
                                                     (short)0, acc0, false, false);
      acc0 = __builtin_amdgcn_wmma_f32_16x16x32_bf16(false, Alo[0][c], false, Bhi,
                                                     (short)0, acc0, false, false);
      acc1 = __builtin_amdgcn_wmma_f32_16x16x32_bf16(false, Ahi[1][c], false, Bhi,
                                                     (short)0, acc1, false, false);
      acc1 = __builtin_amdgcn_wmma_f32_16x16x32_bf16(false, Ahi[1][c], false, Blo,
                                                     (short)0, acc1, false, false);
      acc1 = __builtin_amdgcn_wmma_f32_16x16x32_bf16(false, Alo[1][c], false, Bhi,
                                                     (short)0, acc1, false, false);
    }

    const float es = esq[ct * 16 + n];
    const int   ci = ct * 16 + n;
    #pragma unroll
    for (int v = 0; v < 8; ++v) {
      float d0 = es - 2.0f * acc0[v];
      if (d0 < best[0][v]) { best[0][v] = d0; bidx[0][v] = ci; }  // first-min kept
      float d1 = es - 2.0f * acc1[v];
      if (d1 < best[1][v]) { best[1][v] = d1; bidx[1][v] = ci; }
    }
  }

  // ---- Per-row argmin across the 16 lanes of each half-wave, then gather --
  #pragma unroll
  for (int t = 0; t < 2; ++t) {
    #pragma unroll
    for (int v = 0; v < 8; ++v) {
      float b  = best[t][v];
      int   bi = bidx[t][v];
      #pragma unroll
      for (int off = 8; off >= 1; off >>= 1) {   // stays inside 16-lane group
        float ob = __shfl_xor(b, off, 32);
        int   oi = __shfl_xor(bi, off, 32);
        if (ob < b || (ob == b && oi < bi)) { b = ob; bi = oi; }
      }
      const int r = rowBase + t * 16 + v + 8 * h;
      const int q = bi;
      float* orow = out + (size_t)r * D_DIM;
      #pragma unroll
      for (int j = 0; j < 16; ++j) {             // coalesced 64B stores
        const int d = j * 16 + n;
        orow[d] = emb[(size_t)d * K_CODES + q];  // exact fp32 codebook values
      }
      if (n == 0) atomicAdd(&counts[q], 1.0f);
    }
  }
}

// ---------------------------------------------------------------------------
// Kernel 3: perplexity = exp(-sum(p * log(p + 1e-10))) over usage histogram.
// ---------------------------------------------------------------------------
__global__ void vq_perplexity_kernel(const float* __restrict__ counts,
                                     float* __restrict__ out_scalar,
                                     int n_rows) {
  __shared__ float red[256];
  float s = 0.0f;
  const float inv = 1.0f / (float)n_rows;
  for (int k = threadIdx.x; k < K_CODES; k += blockDim.x) {
    float p = counts[k] * inv;
    s += p * logf(p + 1e-10f);
  }
  red[threadIdx.x] = s;
  __syncthreads();
  for (int st = 128; st > 0; st >>= 1) {
    if (threadIdx.x < st) red[threadIdx.x] += red[threadIdx.x + st];
    __syncthreads();
  }
  if (threadIdx.x == 0) out_scalar[0] = expf(-red[0]);
}

// ---------------------------------------------------------------------------
extern "C" void kernel_launch(void* const* d_in, const int* in_sizes, int n_in,
                              void* d_out, int out_size, void* d_ws, size_t ws_size,
                              hipStream_t stream) {
  const float* x   = (const float*)d_in[0];   // [N, 256] flattened latents
  const float* emb = (const float*)d_in[1];   // [256, 4096]
  float* out = (float*)d_out;                 // [N*256 quantized] + [1 perplexity]

  const int n_rows = in_sizes[0] / D_DIM;     // 65536

  // Workspace layout (32B-aligned chunks):
  //   [0, 2MB)        bhi_pack  (1M x ushort, WMMA-swizzled bf16 hi)
  //   [2MB, 4MB)      blo_pack  (1M x ushort, WMMA-swizzled bf16 lo)
  //   [4MB, +16KB)    esq       (4096 floats)
  //   [.., +16KB)     counts    (4096 floats)
  unsigned short* bhi_pack = (unsigned short*)d_ws;
  unsigned short* blo_pack = bhi_pack + PACK_ELEMS;
  float* esq    = (float*)(blo_pack + PACK_ELEMS);
  float* counts = esq + K_CODES;

  vq_pack_kernel<<<PACK_ELEMS / 256, 256, 0, stream>>>(emb, bhi_pack, blo_pack);
  vq_norms_kernel<<<K_CODES / 256, 256, 0, stream>>>(emb, esq, counts);

  const int strips = (n_rows + 31) / 32;
  const int blocks = (strips + 7) / 8;
  vq_wmma_kernel<<<blocks, 256, 0, stream>>>(x, bhi_pack, blo_pack, esq, emb,
                                             out, counts, n_rows);

  vq_perplexity_kernel<<<1, 256, 0, stream>>>(counts, out + (size_t)n_rows * D_DIM,
                                              n_rows);
}
